// MultiHeadedAttention_89885075570737
// MI455X (gfx1250) — matrix operand order repaired and checked by reading, not compiled, not run on hardware
//
#include <hip/hip_runtime.h>

// ---------------------------------------------------------------------------
// MultiHeadedAttention for MI455X (gfx1250, wave32, WMMA 16x16x32 f16->f32).
// N=4, S=2048, D_MODEL=512, H=8, HEAD_DIM=64.
// Compute-bound (~38 GFLOP vs ~66 MB HBM traffic) -> all matmuls via v_wmma.
// K tiles: GLOBAL_LOAD_ASYNC_TO_LDS_B128 (ASYNCcnt / s_wait_asynccnt).
// V tiles: TENSOR_LOAD_TO_LDS 2D descriptor (TDM, TENSORcnt / s_wait_tensorcnt).
// ---------------------------------------------------------------------------

typedef __attribute__((ext_vector_type(8)))  _Float16 v8h;
typedef __attribute__((ext_vector_type(16))) _Float16 v16h;
typedef __attribute__((ext_vector_type(8)))  float    v8f;
typedef __attribute__((ext_vector_type(4)))  unsigned u32x4;
typedef __attribute__((ext_vector_type(8)))  unsigned u32x8;

#define NB 4
#define NH 8
#define SEQ 2048
#define HD 64
#define DM 512

#define WMMA_F16(a, b, c) \
  __builtin_amdgcn_wmma_f32_16x16x32_f16(false, (a), false, (b), (short)0, (c), false, false)

union U16h { v16h v; v8h h2[2]; };

// Async copy of 16 bytes global -> LDS (VDST = LDS byte address, GV mode).
// Low 32 bits of a flat pointer to LDS are the LDS byte offset (ISA 10.2).
static __device__ __forceinline__ void async_copy_b128(const _Float16* g, _Float16* l) {
  unsigned lds = (unsigned)(unsigned long long)(void*)l;
  unsigned long long ga = (unsigned long long)g;
  asm volatile("global_load_async_to_lds_b128 %0, %1, off" :: "v"(lds), "v"(ga) : "memory");
}

// TDM: one 2D f16 tile (tileD0 contiguous elems x tileD1 lines, line stride
// strideD0 elems) global -> LDS. D# group0/group1 packed per ISA ch.8.3/8.4,
// VADDR2/3 omitted (<=2D). Issued once per wave; tracked on TENSORcnt.
static __device__ __forceinline__ void tdm_load_2d_f16(const _Float16* g, unsigned ldsOff,
                                                       unsigned tensorD0, unsigned tensorD1,
                                                       unsigned tileD0, unsigned tileD1,
                                                       unsigned strideD0) {
  unsigned long long ga = (unsigned long long)g;
  u32x4 g0;
  g0[0] = 1u;                                              // count=1 (valid), user mode
  g0[1] = ldsOff;                                          // lds_addr   [63:32]
  g0[2] = (unsigned)(ga & 0xffffffffu);                    // global_addr[95:64]
  g0[3] = (unsigned)((ga >> 32) & 0x01ffffffu) | (2u << 30); // addr[56:32] | type=2
  u32x8 g1;
  g1[0] = 1u << 16;                                        // data_size=1 (2 bytes)
  g1[1] = (tensorD0 & 0xffffu) << 16;                      // tensor_dim0 lo @ [63:48]
  g1[2] = ((tensorD0 >> 16) & 0xffffu) | ((tensorD1 & 0xffffu) << 16);
  g1[3] = ((tensorD1 >> 16) & 0xffffu) | (tileD0 << 16);   // | tile_dim0 @ [127:112]
  g1[4] = tileD1 & 0xffffu;                                // tile_dim1, tile_dim2=0
  g1[5] = strideD0;                                        // tensor_dim0_stride lo
  g1[6] = 0u;                                              // stride hi | dim1_stride lo
  g1[7] = 0u;
  asm volatile("tensor_load_to_lds %0, %1" :: "s"(g0), "s"(g1) : "memory");
}

// -------------------------------------------------------------------- convert
__global__ __launch_bounds__(256) void cvt_f32_to_f16(const float* __restrict__ src,
                                                      _Float16* __restrict__ dst, int n) {
  int i = blockIdx.x * 256 + threadIdx.x;
  if (i < n) dst[i] = (_Float16)src[i];
}

// ----------------------------------------------------------------- projection
// X[n,s,DM] (f32), head slice h*64.. ; W16 is 64x64 f16 (row-major, == W).
// out = X_head(16x64) @ W^T via WMMA: B column n0+l is W row n0+l (contiguous).
// transposed==0: store row-major [ (n*H+h)*S + s ][ d ]   (Q, K)
// transposed==1: store           [ (n*H+h)*D + d ][ s ]   (V)
__global__ __launch_bounds__(32)
void proj_kernel(const float* __restrict__ X, const _Float16* __restrict__ W16,
                 _Float16* __restrict__ out, int transposed) {
  const int lane = threadIdx.x;
  const int tile = blockIdx.x;       // 0..NB*SEQ/16-1
  const int h    = blockIdx.y;       // 0..NH-1
  const int tt0  = tile * 16;
  const int n    = tt0 / SEQ;
  const int s0   = tt0 % SEQ;        // tiles never cross batch boundary
  const int col  = lane & 15;
  const int aoff = (lane < 16) ? 0 : 8;
  const int boff = (lane < 16) ? 0 : 16;

  // A vectors: row = token tt0+col, K = head dims (f32 -> f16)
  const float* xrow = X + (size_t)(tt0 + col) * DM + h * HD;
  U16h a0, a1;
#pragma unroll
  for (int e = 0; e < 8; ++e) {
    a0.v[e]     = (_Float16)xrow[boff + e];
    a0.v[e + 8] = (_Float16)xrow[boff + 8 + e];
    a1.v[e]     = (_Float16)xrow[32 + boff + e];
    a1.v[e + 8] = (_Float16)xrow[32 + boff + 8 + e];
  }

#pragma unroll
  for (int j = 0; j < 4; ++j) {
    const _Float16* wrow = W16 + (size_t)(j * 16 + col) * HD;
    v16h b0 = *(const v16h*)(wrow + boff);        // K = 0..31 chunk
    v16h b1 = *(const v16h*)(wrow + 32 + boff);   // K = 32..63 chunk
    v8f c = {};
    c = WMMA_F16(a0.v, b0, c);
    c = WMMA_F16(a1.v, b1, c);

    const int dim = j * 16 + col;
    if (!transposed) {
      size_t base = ((size_t)(n * NH + h) * SEQ + s0 + ((lane < 16) ? 0 : 8)) * HD + dim;
#pragma unroll
      for (int r = 0; r < 8; ++r) out[base + (size_t)r * HD] = (_Float16)c[r];
    } else {
      v8h pk;
#pragma unroll
      for (int r = 0; r < 8; ++r) pk[r] = (_Float16)c[r];
      size_t base = ((size_t)(n * NH + h) * HD + dim) * SEQ + s0 + ((lane < 16) ? 0 : 8);
      *(v8h*)(out + base) = pk;   // 8 token-consecutive values, one b128 store
    }
  }
}

// ----------------------------------------------------------- flash attention
// Block = 4 waves = 64 q rows for one (n,h). Each 32-token K/V tile is staged
// once into LDS (double buffered): K via per-lane async b128, V via one TDM
// 2D-tile descriptor from wave 0.
__global__ __launch_bounds__(128)
void attn_kernel(const _Float16* __restrict__ Qh, const _Float16* __restrict__ Kh,
                 const _Float16* __restrict__ Vt, const int* __restrict__ mask,
                 _Float16* __restrict__ A16) {
  __shared__ _Float16 ktile[2][32][HD];    // [buf][kv token][dim]     8 KB
  __shared__ _Float16 vtile[2][HD][32];    // [buf][dim][kv token]     8 KB
  __shared__ _Float16 pl[4][16][40];       // per-wave P tile, padded  5 KB

  const int tid  = threadIdx.x;
  const int lane = tid & 31;
  const int w    = tid >> 5;
  const int h  = blockIdx.y;
  const int n  = blockIdx.z;
  const int q0 = blockIdx.x * 64 + w * 16;
  const int col  = lane & 15;
  const int aoff = (lane < 16) ? 0 : 8;
  const int boff = (lane < 16) ? 0 : 16;
  const size_t headBase = (size_t)(n * NH + h) * SEQ * HD;   // Qh / Kh
  const _Float16* Kbase = Kh + headBase;                     // [seq][64]
  const _Float16* Vbase = Vt + (size_t)(n * NH + h) * HD * SEQ;  // [dim][seq]

  // Q in A layout (f16, contiguous 16B chunks)
  const _Float16* qrow = Qh + headBase + (size_t)(q0 + col) * HD;
  U16h qa0, qa1;
  qa0.h2[0] = *(const v8h*)(qrow + boff);
  qa0.h2[1] = *(const v8h*)(qrow + boff + 8);
  qa1.h2[0] = *(const v8h*)(qrow + 32 + boff);
  qa1.h2[1] = *(const v8h*)(qrow + 32 + boff + 8);

  v8f o0 = {}, o1 = {}, o2 = {}, o3 = {};
  float rm[8], rs[8];
#pragma unroll
  for (int r = 0; r < 8; ++r) { rm[r] = -3.0e38f; rs[r] = 0.0f; }

  const int* mrow = mask + (size_t)n * SEQ;

  // K tile: contiguous 4 KB -> 256 b128 async transfers over 128 threads
  // (2 ASYNCcnt increments per wave per tile).
  auto issue_k = [&](int kv, int buf) {
#pragma unroll
    for (int i = 0; i < 2; ++i) {
      int c = tid + i * 128;                 // 16B chunk id, 0..255
      async_copy_b128(Kbase + (size_t)kv * HD + c * 8, &ktile[buf][0][0] + c * 8);
    }
  };
  // V tile: 64 lines x 32 elems, line stride SEQ -> one TDM descriptor
  // (1 TENSORcnt increment on wave 0).
  auto issue_v = [&](int kv, int buf) {
    if (w == 0) {
      unsigned ldsOff = (unsigned)(unsigned long long)(void*)&vtile[buf][0][0];
      tdm_load_2d_f16(Vbase + kv, ldsOff,
                      /*tensorD0=*/SEQ, /*tensorD1=*/HD,
                      /*tileD0=*/32, /*tileD1=*/HD, /*strideD0=*/SEQ);
    }
  };

  issue_k(0, 0);
  issue_v(0, 0);
  int buf = 0;

  for (int kv = 0; kv < SEQ; kv += 32) {
    if (kv + 32 < SEQ) {
      issue_k(kv + 32, buf ^ 1);
      issue_v(kv + 32, buf ^ 1);
      asm volatile("s_wait_asynccnt 0x2" ::: "memory");      // prev K tile landed
      if (w == 0) __builtin_amdgcn_s_wait_tensorcnt(1);      // prev V tile landed
    } else {
      asm volatile("s_wait_asynccnt 0x0" ::: "memory");
      if (w == 0) __builtin_amdgcn_s_wait_tensorcnt(0);
    }
    __syncthreads();   // tile[buf] visible to all 4 waves

    // ---- scores: two 16x16 tiles, K(head dim)=64 in two WMMA steps (LDS B)
    const _Float16* k0p = &ktile[buf][col][boff];
    const _Float16* k1p = &ktile[buf][16 + col][boff];
    v16h b00 = *(const v16h*)(k0p);
    v16h b01 = *(const v16h*)(k0p + 32);
    v16h b10 = *(const v16h*)(k1p);
    v16h b11 = *(const v16h*)(k1p + 32);
    v8f c0 = {}, c1 = {};
    c0 = WMMA_F16(qa0.v, b00, c0);
    c0 = WMMA_F16(qa1.v, b01, c0);
    c1 = WMMA_F16(qa0.v, b10, c1);
    c1 = WMMA_F16(qa1.v, b11, c1);

    // ---- masked_fill(-1e20) then *1/sqrt(64)  (reference order)
    const bool m0 = mrow[kv + col] != 0;
    const bool m1 = mrow[kv + 16 + col] != 0;
#pragma unroll
    for (int r = 0; r < 8; ++r) {
      c0[r] = (m0 ? c0[r] : -1.0e20f) * 0.125f;
      c1[r] = (m1 ? c1[r] : -1.0e20f) * 0.125f;
    }

    // ---- online softmax: row stats live across each 16-lane half
    float pmax[8];
#pragma unroll
    for (int r = 0; r < 8; ++r) pmax[r] = fmaxf(c0[r], c1[r]);
#pragma unroll
    for (int m = 1; m < 16; m <<= 1)
#pragma unroll
      for (int r = 0; r < 8; ++r) pmax[r] = fmaxf(pmax[r], __shfl_xor(pmax[r], m, 16));

    float alpha[8];
#pragma unroll
    for (int r = 0; r < 8; ++r) {
      float mnew = fmaxf(rm[r], pmax[r]);
      alpha[r] = __expf(rm[r] - mnew);
      rm[r] = mnew;
    }
    float psum[8];
#pragma unroll
    for (int r = 0; r < 8; ++r) {
      c0[r] = __expf(c0[r] - rm[r]);
      c1[r] = __expf(c1[r] - rm[r]);
      psum[r] = c0[r] + c1[r];
    }
#pragma unroll
    for (int m = 1; m < 16; m <<= 1)
#pragma unroll
      for (int r = 0; r < 8; ++r) psum[r] += __shfl_xor(psum[r], m, 16);
#pragma unroll
    for (int r = 0; r < 8; ++r) {
      rs[r] = rs[r] * alpha[r] + psum[r];
      o0[r] *= alpha[r]; o1[r] *= alpha[r]; o2[r] *= alpha[r]; o3[r] *= alpha[r];
    }

    // ---- P: C layout -> per-wave LDS tile -> A layout
    const int prow = (lane < 16) ? 0 : 8;
#pragma unroll
    for (int r = 0; r < 8; ++r) {
      pl[w][prow + r][col]      = (_Float16)c0[r];
      pl[w][prow + r][16 + col] = (_Float16)c1[r];
    }
    __syncthreads();   // uniform; also covers per-wave P store->load ordering
    U16h pa;
    pa.h2[0] = *(const v8h*)(&pl[w][col][boff]);
    pa.h2[1] = *(const v8h*)(&pl[w][col][boff + 8]);

    // ---- O += P(16x32) @ V(32x64), V tile transposed in LDS
    v16h vb0 = *(const v16h*)(&vtile[buf][col][boff]);
    v16h vb1 = *(const v16h*)(&vtile[buf][16 + col][boff]);
    v16h vb2 = *(const v16h*)(&vtile[buf][32 + col][boff]);
    v16h vb3 = *(const v16h*)(&vtile[buf][48 + col][boff]);
    o0 = WMMA_F16(pa.v, vb0, o0);
    o1 = WMMA_F16(pa.v, vb1, o1);
    o2 = WMMA_F16(pa.v, vb2, o2);
    o3 = WMMA_F16(pa.v, vb3, o3);

    __syncthreads();   // WAR: tile[buf^1]/pl reused by next iteration's issue
    buf ^= 1;
  }

  // ---- normalize and store [n, s, h*64+d] as f16 (FC input)
  float inv[8];
#pragma unroll
  for (int r = 0; r < 8; ++r) inv[r] = 1.0f / rs[r];
  const int srow = q0 + ((lane < 16) ? 0 : 8);
#pragma unroll
  for (int r = 0; r < 8; ++r) {
    size_t base = ((size_t)n * SEQ + srow + r) * DM + h * HD + col;
    A16[base]      = (_Float16)(o0[r] * inv[r]);
    A16[base + 16] = (_Float16)(o1[r] * inv[r]);
    A16[base + 32] = (_Float16)(o2[r] * inv[r]);
    A16[base + 48] = (_Float16)(o3[r] * inv[r]);
  }
}

// ------------------------------------------------------------------ final FC
// out[t, o] = sum_k A16[t,k] * Wfc[o,k] + bfc[o]   (B column o = Wfc row o)
__global__ __launch_bounds__(32)
void fc_kernel(const _Float16* __restrict__ A16, const _Float16* __restrict__ Wfc16,
               const float* __restrict__ bfc, float* __restrict__ out) {
  const int lane = threadIdx.x;
  const int mt = blockIdx.x;      // token tile 0..511
  const int nt = blockIdx.y;      // out-col tile 0..31
  const int col  = lane & 15;
  const int aoff = (lane < 16) ? 0 : 8;
  const int boff = (lane < 16) ? 0 : 16;

  const _Float16* arow = A16 + (size_t)(mt * 16 + col) * DM;
  const _Float16* wrow = Wfc16 + (size_t)(nt * 16 + col) * DM;
  v8f c = {};
#pragma unroll
  for (int kb = 0; kb < DM; kb += 32) {
    if (kb + 32 < DM) __builtin_prefetch((const void*)(wrow + kb + 32), 0, 0);
    U16h a;
    a.h2[0] = *(const v8h*)(arow + kb + boff);
    a.h2[1] = *(const v8h*)(arow + kb + boff + 8);
    v16h b = *(const v16h*)(wrow + kb + boff);
    c = WMMA_F16(a.v, b, c);
  }
  const int dim = nt * 16 + col;
  const float bias = bfc[dim];
  const size_t r0 = (size_t)(mt * 16 + ((lane < 16) ? 0 : 8));
#pragma unroll
  for (int r = 0; r < 8; ++r) out[(r0 + r) * DM + dim] = c[r] + bias;
}

// ----------------------------------------------------------------------------
extern "C" void kernel_launch(void* const* d_in, const int* in_sizes, int n_in,
                              void* d_out, int out_size, void* d_ws, size_t ws_size,
                              hipStream_t stream) {
  const float* value = (const float*)d_in[0];
  const float* key   = (const float*)d_in[1];
  const float* query = (const float*)d_in[2];
  const int*   mask  = (const int*)d_in[3];
  const float* Wv    = (const float*)d_in[4];
  const float* Wk    = (const float*)d_in[5];
  const float* Wq    = (const float*)d_in[6];
  const float* Wfc   = (const float*)d_in[7];
  const float* bfc   = (const float*)d_in[8];
  float* out = (float*)d_out;

  const size_t perHead = (size_t)NB * NH * SEQ * HD;     // 4,194,304
  _Float16* Qh    = (_Float16*)d_ws;
  _Float16* Kh    = Qh + perHead;
  _Float16* Vt    = Kh + perHead;
  _Float16* A16   = Vt + perHead;
  _Float16* Wfc16 = A16 + (size_t)NB * SEQ * DM;
  _Float16* Wv16  = Wfc16 + (size_t)DM * DM;
  _Float16* Wk16  = Wv16 + HD * HD;
  _Float16* Wq16  = Wk16 + HD * HD;
  const size_t need = (size_t)(Wq16 + HD * HD - Qh) * sizeof(_Float16);
  if (ws_size < need) return;

  // 1) weights -> f16
  cvt_f32_to_f16<<<(DM * DM + 255) / 256, 256, 0, stream>>>(Wfc, Wfc16, DM * DM);
  cvt_f32_to_f16<<<(HD * HD + 255) / 256, 256, 0, stream>>>(Wv, Wv16, HD * HD);
  cvt_f32_to_f16<<<(HD * HD + 255) / 256, 256, 0, stream>>>(Wk, Wk16, HD * HD);
  cvt_f32_to_f16<<<(HD * HD + 255) / 256, 256, 0, stream>>>(Wq, Wq16, HD * HD);

  // 2) projections (Q,K row-major; V transposed per head)
  dim3 pgrid(NB * SEQ / 16, NH);
  proj_kernel<<<pgrid, 32, 0, stream>>>(query, Wq16, Qh, 0);
  proj_kernel<<<pgrid, 32, 0, stream>>>(key,   Wk16, Kh, 0);
  proj_kernel<<<pgrid, 32, 0, stream>>>(value, Wv16, Vt, 1);

  // 3) flash attention (4 waves / block, async + TDM double-buffered tiles)
  dim3 agrid(SEQ / 64, NH, NB);
  attn_kernel<<<agrid, 128, 0, stream>>>(Qh, Kh, Vt, mask, A16);

  // 4) output projection + bias
  dim3 fgrid(NB * SEQ / 16, DM / 16);
  fc_kernel<<<fgrid, 32, 0, stream>>>(A16, Wfc16, bfc, out);
}
